// ColorAttentionGNN_72748156060194
// MI455X (gfx1250) — compile-verified, matrix operand order checked by reading, not run on hardware
//
#include <hip/hip_runtime.h>
#include <hip/hip_bf16.h>

// ---------------- types for WMMA ----------------
typedef __bf16 v16bf __attribute__((ext_vector_type(16)));
typedef float  v8f   __attribute__((ext_vector_type(8)));
typedef int    v4i   __attribute__((ext_vector_type(4)));

#define AS1 __attribute__((address_space(1)))
#define AS3 __attribute__((address_space(3)))

#if __has_builtin(__builtin_amdgcn_global_load_async_to_lds_b128) && \
    __has_builtin(__builtin_amdgcn_s_wait_asynccnt)
#define HAVE_ASYNC_LDS 1
#else
#define HAVE_ASYNC_LDS 0
#endif

// Let the backend use native bf16 converts (v_cvt_pk_bf16_f32 on gfx1250).
__device__ __forceinline__ unsigned short f32_to_bf16(float f) {
    __bf16 b = (__bf16)f;
    return __builtin_bit_cast(unsigned short, b);
}
__device__ __forceinline__ unsigned int pack_bf16x2(float lo, float hi) {
    return (unsigned int)f32_to_bf16(lo) | ((unsigned int)f32_to_bf16(hi) << 16);
}

#define NNODES 10000
#define NEDGES 80000
#define ETOT   90000          // edges + self loops
#define NEGS   0.2f
#define EPSB   1e-5f
#define HCMAX  4096
#define HEMB   1755
#define KP0    1792           // HEMB padded to multiple of 64

// ---------------- fill kernels ----------------
__global__ void fill_f32(float* p, long long n, float v) {
    long long i = (long long)blockIdx.x * blockDim.x + threadIdx.x;
    if (i < n) p[i] = v;
}
__global__ void fill_i32(int* p, long long n, int v) {
    long long i = (long long)blockIdx.x * blockDim.x + threadIdx.x;
    if (i < n) p[i] = v;
}

// ---------------- embedding: build h0 (N x 1792, zero-padded past 1755) ----------------
__global__ void embed_kernel(const float* __restrict__ x,
                             const float* __restrict__ emb_layer,   // 4 x 250
                             const float* __restrict__ emb_rel,     // 11 x 250
                             const float* __restrict__ emb_color,   // 256 x 85
                             float* __restrict__ h) {
    long long idx = (long long)blockIdx.x * blockDim.x + threadIdx.x;
    const long long total = (long long)NNODES * KP0;
    if (idx >= total) return;
    int n = (int)(idx / KP0);
    int j = (int)(idx - (long long)n * KP0);
    const float* xr = x + (long long)n * 1005;
    float v = 0.0f;
    if (j < 250) {
        int c = (int)xr[0];
        v = emb_layer[c * 250 + j];
    } else if (j < 1250) {
        v = xr[1 + (j - 250)];
    } else if (j < 1500) {
        int c = __float2int_rn(xr[1001] * 10.0f);
        v = emb_rel[c * 250 + (j - 1250)];
    } else if (j < HEMB) {
        int jj = j - 1500;
        int which = jj / 85, o = jj - which * 85;
        int c = (int)xr[1002 + which];
        v = emb_color[c * 85 + o];
    }
    h[idx] = v;
}

// ---------------- self loops: deg & sum(edge_attr) per dst ----------------
__global__ void selfloop_deg_kernel(const int* __restrict__ ei,
                                    const float* __restrict__ ea,
                                    float* __restrict__ ea_full,
                                    float* __restrict__ deg,
                                    float* __restrict__ sumea) {
    int e = blockIdx.x * blockDim.x + threadIdx.x;
    if (e >= NEDGES) return;
    int d = ei[NEDGES + e];
    float a = ea[e];
    ea_full[e] = a;
    atomicAdd(&deg[d], 1.0f);
    atomicAdd(&sumea[d], a);
}
__global__ void selfloop_mean_kernel(const float* __restrict__ deg,
                                     const float* __restrict__ sumea,
                                     float* __restrict__ ea_full) {
    int n = blockIdx.x * blockDim.x + threadIdx.x;
    if (n >= NNODES) return;
    float dg = deg[n];
    ea_full[NEDGES + n] = sumea[n] / (dg < 1.0f ? 1.0f : dg);
}

// ---------------- weight transpose + f32 -> bf16, rows padded to Kpad ----------------
// W is (K x Nn) row-major; produce Wt (Nn x Kpad) bf16, zero-filled for k >= K.
__global__ void convert_weight_kernel(const float* __restrict__ W,
                                      unsigned short* __restrict__ Wt,
                                      int K, int Kpad, int Nn) {
    long long idx = (long long)blockIdx.x * blockDim.x + threadIdx.x;
    long long total = (long long)Kpad * Nn;
    if (idx >= total) return;
    int n = (int)(idx / Kpad);
    int k = (int)(idx - (long long)n * Kpad);
    Wt[idx] = (k < K) ? f32_to_bf16(W[(long long)k * Nn + n]) : (unsigned short)0;
}

// ---------------- WMMA bf16 GEMM, ping-pong LDS + prefetch ----------------
// C(M x Nn) = A(M x Kpad, f32, zero-padded) * Bt(Nn x Kpad, bf16)^T
#define BM 128
#define BN 128
#define BK 64

__global__ __launch_bounds__(256)
void gemm_bf16_wmma(const float* __restrict__ A,
                    const unsigned short* __restrict__ Bt,
                    float* __restrict__ C,
                    int M, int Kpad, int Nn) {
    __shared__ __attribute__((aligned(128))) unsigned short As[2][BM * BK]; // 2 x 16 KB
    __shared__ __attribute__((aligned(128))) unsigned short Bs[2][BN * BK]; // 2 x 16 KB

    const int tid  = threadIdx.x;
    const int lane = tid & 31;
    const int wid  = tid >> 5;          // 8 waves
    const int wm   = wid & 1;           // 2 wave rows  -> 64 M each
    const int wn   = wid >> 1;          // 4 wave cols  -> 32 N each
    const int bm   = blockIdx.y * BM;
    const int bn   = blockIdx.x * BN;
    const int half = lane >> 4;
    const int lr   = lane & 15;
    const bool bInterior = (bn + BN <= Nn);

    v8f acc[4][2] = {};
    float4 ar[8];

    auto loadA = [&](int k0) {
        #pragma unroll
        for (int it = 0; it < 8; ++it) {
            int t = tid + it * 256;     // 0..2047
            int r = t >> 4;             // row, 16 float4 per row
            int q = t & 15;             // float4 index within row
            int gr = bm + r;
            ar[it] = make_float4(0.f, 0.f, 0.f, 0.f);
            if (gr < M) ar[it] = *(const float4*)&A[(long long)gr * Kpad + k0 + q * 4];
        }
    };
    auto storeA = [&](int p) {
        #pragma unroll
        for (int it = 0; it < 8; ++it) {
            int t = tid + it * 256;
            int r = t >> 4, q = t & 15;
            uint2 pk;
            pk.x = pack_bf16x2(ar[it].x, ar[it].y);
            pk.y = pack_bf16x2(ar[it].z, ar[it].w);
            *(uint2*)&As[p][r * BK + q * 4] = pk;
        }
    };
    auto issueB = [&](int p, int k0) {
        if (bInterior) {
#if HAVE_ASYNC_LDS
            #pragma unroll
            for (int it = 0; it < 4; ++it) {
                int t = tid + it * 256; // 0..1023 ; 8 x 16B chunks per row
                int r = t >> 3, q = t & 7;
                unsigned short* g = const_cast<unsigned short*>(
                    &Bt[(long long)(bn + r) * Kpad + k0 + q * 8]);
                __builtin_amdgcn_global_load_async_to_lds_b128(
                    (AS1 v4i*)(void*)g,
                    (AS3 v4i*)(void*)&Bs[p][r * BK + q * 8], 0, 0);
            }
#else
            #pragma unroll
            for (int it = 0; it < 4; ++it) {
                int t = tid + it * 256;
                int r = t >> 3, q = t & 7;
                uint4 v = *(const uint4*)&Bt[(long long)(bn + r) * Kpad + k0 + q * 8];
                *(uint4*)&Bs[p][r * BK + q * 8] = v;
            }
#endif
        } else {
            // boundary block in N (only the last layer): guarded elementwise
            for (int t = tid; t < BN * BK; t += 256) {
                int r = t / BK, c = t - (t / BK) * BK;
                int gn = bn + r;
                Bs[p][t] = (gn < Nn) ? Bt[(long long)gn * Kpad + k0 + c]
                                     : (unsigned short)0;
            }
        }
    };

    // prologue: first A tile into registers
    loadA(0);

    int p = 0;
    for (int k0 = 0; k0 < Kpad; k0 += BK) {
        // stage current tile into buffer p
        storeA(p);
        issueB(p, k0);
        // prefetch next A tile into registers (overlaps with wait + compute)
        if (k0 + BK < Kpad) loadA(k0 + BK);
#if HAVE_ASYNC_LDS
        if (bInterior) __builtin_amdgcn_s_wait_asynccnt(0);
#endif
        __syncthreads();   // single barrier per iteration (ping-pong makes this safe)

        // ---- compute: 2 WMMA k-slices of 32 ----
        #pragma unroll
        for (int kk = 0; kk < BK; kk += 32) {
            v16bf af[4], bfr[2];
            #pragma unroll
            for (int i = 0; i < 4; ++i)
                af[i] = *(const v16bf*)&As[p][(wm * 64 + i * 16 + lr) * BK + kk + half * 16];
            #pragma unroll
            for (int j = 0; j < 2; ++j)
                bfr[j] = *(const v16bf*)&Bs[p][(wn * 32 + j * 16 + lr) * BK + kk + half * 16];
            #pragma unroll
            for (int i = 0; i < 4; ++i)
                #pragma unroll
                for (int j = 0; j < 2; ++j)
                    acc[i][j] = __builtin_amdgcn_wmma_f32_16x16x32_bf16(
                        false, af[i], false, bfr[j], (short)0, acc[i][j],
                        false, false);
        }
        p ^= 1;
    }

    // C/D layout: VGPR r -> M = r + 8*half ; N = lr
    #pragma unroll
    for (int i = 0; i < 4; ++i)
        #pragma unroll
        for (int j = 0; j < 2; ++j)
            #pragma unroll
            for (int r = 0; r < 8; ++r) {
                int row = bm + wm * 64 + i * 16 + r + 8 * half;
                int col = bn + wn * 32 + j * 16 + lr;
                if (row < M && col < Nn)
                    C[(long long)row * Nn + col] = acc[i][j][r];
            }
}

// ---------------- attention scores: one wave per (edge, head) ----------------
__global__ __launch_bounds__(256)
void score_kernel(const float* __restrict__ xl, const float* __restrict__ xr,
                  const int* __restrict__ ei, const float* __restrict__ eaf,
                  const float* __restrict__ We, const float* __restrict__ att,
                  float* __restrict__ score, int* __restrict__ smax,
                  int H, int Cc) {
    int gw = blockIdx.x * (blockDim.x >> 5) + (threadIdx.x >> 5);
    int lane = threadIdx.x & 31;
    if (gw >= ETOT * H) return;
    int e = gw / H, h = gw - e * H;
    int s, d;
    if (e < NEDGES) { s = ei[e]; d = ei[NEDGES + e]; }
    else            { s = d = e - NEDGES; }
    float ea = eaf[e];
    const long long HC = (long long)H * Cc;
    const float* xls = xl + (long long)s * HC + (long long)h * Cc;
    const float* xrd = xr + (long long)d * HC + (long long)h * Cc;
    const float* Weh = We + h * Cc;
    const float* ath = att + h * Cc;
    float p = 0.0f;
    for (int c = lane; c < Cc; c += 32) {
        float v = xls[c] + xrd[c] + ea * Weh[c];
        v = v > 0.0f ? v : NEGS * v;
        p += v * ath[c];
    }
    #pragma unroll
    for (int o = 16; o > 0; o >>= 1) p += __shfl_xor(p, o, 32);
    if (lane == 0) {
        score[gw] = p;
        int enc = __float_as_int(p);
        enc = enc >= 0 ? enc : (enc ^ 0x7fffffff);   // orderable float encoding
        atomicMax(&smax[d * H + h], enc);
    }
}

// ---------------- exp(score - segmax) and denominator ----------------
__global__ void softmax_edge_kernel(const float* __restrict__ score,
                                    const int* __restrict__ ei,
                                    const int* __restrict__ smax,
                                    float* __restrict__ exb,
                                    float* __restrict__ den, int H) {
    int gw = blockIdx.x * blockDim.x + threadIdx.x;
    if (gw >= ETOT * H) return;
    int e = gw / H, h = gw - e * H;
    int d = (e < NEDGES) ? ei[NEDGES + e] : (e - NEDGES);
    int enc = smax[d * H + h];
    float sm = (enc >= 0) ? __int_as_float(enc) : __int_as_float(enc ^ 0x7fffffff);
    float ex = __expf(score[gw] - sm);
    exb[gw] = ex;
    atomicAdd(&den[d * H + h], ex);
}

// ---------------- scatter aggregate: out[d] += alpha * xl[s], 4 elems/thread ----------------
__global__ void aggregate_kernel(const float* __restrict__ xl,
                                 const int* __restrict__ ei,
                                 const float* __restrict__ exb,
                                 const float* __restrict__ den,
                                 float* __restrict__ out, int H, int Cc) {
    const long long HC = (long long)H * Cc;
    long long idx = (long long)blockIdx.x * blockDim.x + threadIdx.x;
    long long total4 = (long long)ETOT * HC / 4;
    if (idx >= total4) return;
    long long base = idx * 4;
    int e = (int)(base / HC);
    int r = (int)(base - (long long)e * HC);   // multiple of 4; Cc % 4 == 0 -> same head
    int h = r / Cc;
    int s, d;
    if (e < NEDGES) { s = ei[e]; d = ei[NEDGES + e]; }
    else            { s = d = e - NEDGES; }
    float alpha = exb[(long long)e * H + h] / den[d * H + h];
    float4 v = *(const float4*)&xl[(long long)s * HC + r];
    float* o = &out[(long long)d * HC + r];
    atomicAdd(o + 0, alpha * v.x);
    atomicAdd(o + 1, alpha * v.y);
    atomicAdd(o + 2, alpha * v.z);
    atomicAdd(o + 3, alpha * v.w);
}

// ---------------- bias (+ optional BN + ELU) epilogue ----------------
__global__ void bias_bn_elu_kernel(const float* __restrict__ out,
                                   const float* __restrict__ b,
                                   const float* __restrict__ gamma,
                                   const float* __restrict__ beta,
                                   const float* __restrict__ mean,
                                   const float* __restrict__ var,
                                   float* __restrict__ hnext,
                                   int HC, int do_bnelu) {
    long long idx = (long long)blockIdx.x * blockDim.x + threadIdx.x;
    long long total = (long long)NNODES * HC;
    if (idx >= total) return;
    int j = (int)(idx % HC);
    float v = out[idx] + b[j];
    if (do_bnelu) {
        v = gamma[j] * (v - mean[j]) * rsqrtf(var[j] + EPSB) + beta[j];
        v = v > 0.0f ? v : expm1f(v);
    }
    hnext[idx] = v;
}

// ---------------- final linear: (N x 64) @ (64 x 3) + b ----------------
__global__ void final_lin_kernel(const float* __restrict__ h,
                                 const float* __restrict__ W,
                                 const float* __restrict__ b,
                                 float* __restrict__ y) {
    int n = blockIdx.x * blockDim.x + threadIdx.x;
    if (n >= NNODES) return;
    float a0 = 0.f, a1 = 0.f, a2 = 0.f;
    const float* hn = h + (long long)n * 64;
    #pragma unroll
    for (int k = 0; k < 64; ++k) {
        float hv = hn[k];
        a0 += hv * W[k * 3 + 0];
        a1 += hv * W[k * 3 + 1];
        a2 += hv * W[k * 3 + 2];
    }
    y[n * 3 + 0] = a0 + b[0];
    y[n * 3 + 1] = a1 + b[1];
    y[n * 3 + 2] = a2 + b[2];
}

// ---------------- host orchestration ----------------
static inline long long cdiv(long long a, long long b) { return (a + b - 1) / b; }

extern "C" void kernel_launch(void* const* d_in, const int* in_sizes, int n_in,
                              void* d_out, int out_size, void* d_ws, size_t ws_size,
                              hipStream_t stream) {
    (void)in_sizes; (void)n_in; (void)out_size; (void)ws_size;

    const float* x     = (const float*)d_in[0];
    const int*   ei    = (const int*)  d_in[1];
    const float* eattr = (const float*)d_in[2];
    const float* embL  = (const float*)d_in[3];
    const float* embR  = (const float*)d_in[4];
    const float* embC  = (const float*)d_in[5];
    // gat i: 6 + 5*i : Wl, Wr, We, att, b ; bn i: 26 + 4*i ; lin: 38, 39
    const float* linW = (const float*)d_in[38];
    const float* linB = (const float*)d_in[39];

    const int FIN[4] = {1755, 4096, 2048, 512};
    const int KP [4] = { KP0, 4096, 2048, 512};   // padded K (multiple of 64)
    const int CC [4] = { 512,  256,   64,  64};
    const int HH [4] = {   8,    8,    8,   1};

    // workspace layout
    char* ws = (char*)d_ws;
    size_t off = 0;
    auto take = [&](size_t bytes) { char* p = ws + off; off += (bytes + 255) & ~(size_t)255; return p; };
    float*          hbuf  = (float*)         take((size_t)NNODES * HCMAX * 4);
    float*          xl    = (float*)         take((size_t)NNODES * HCMAX * 4);
    float*          xrout = (float*)         take((size_t)NNODES * HCMAX * 4); // xr, then reused as "out"
    unsigned short* wt    = (unsigned short*)take((size_t)HCMAX * HCMAX * 2);
    float*          eaf   = (float*)         take((size_t)ETOT * 4);
    float*          deg   = (float*)         take((size_t)NNODES * 4);
    float*          sumea = (float*)         take((size_t)NNODES * 4);
    float*          score = (float*)         take((size_t)ETOT * 8 * 4);
    float*          exb   = (float*)         take((size_t)ETOT * 8 * 4);
    int*            smax  = (int*)           take((size_t)NNODES * 8 * 4);
    float*          den   = (float*)         take((size_t)NNODES * 8 * 4);

    const int T = 256;

    // embeddings -> h0 (stride KP0, zero padded)
    embed_kernel<<<cdiv((long long)NNODES * KP0, T), T, 0, stream>>>(x, embL, embR, embC, hbuf);

    // self loops
    fill_f32<<<cdiv(NNODES, T), T, 0, stream>>>(deg,   NNODES, 0.f);
    fill_f32<<<cdiv(NNODES, T), T, 0, stream>>>(sumea, NNODES, 0.f);
    selfloop_deg_kernel<<<cdiv(NEDGES, T), T, 0, stream>>>(ei, eattr, eaf, deg, sumea);
    selfloop_mean_kernel<<<cdiv(NNODES, T), T, 0, stream>>>(deg, sumea, eaf);

    for (int i = 0; i < 4; ++i) {
        const int K    = FIN[i];
        const int Kpad = KP[i];
        const int H    = HH[i];
        const int Cc   = CC[i];
        const int HC   = H * Cc;
        const float* Wl  = (const float*)d_in[6 + 5 * i + 0];
        const float* Wr  = (const float*)d_in[6 + 5 * i + 1];
        const float* We  = (const float*)d_in[6 + 5 * i + 2];
        const float* att = (const float*)d_in[6 + 5 * i + 3];
        const float* bb  = (const float*)d_in[6 + 5 * i + 4];

        dim3 ggrid((unsigned)cdiv(HC, BN), (unsigned)cdiv(NNODES, BM));

        // xl = h @ Wl  (WMMA bf16)
        convert_weight_kernel<<<cdiv((long long)Kpad * HC, T), T, 0, stream>>>(Wl, wt, K, Kpad, HC);
        gemm_bf16_wmma<<<ggrid, 256, 0, stream>>>(hbuf, wt, xl, NNODES, Kpad, HC);
        // xr = h @ Wr
        convert_weight_kernel<<<cdiv((long long)Kpad * HC, T), T, 0, stream>>>(Wr, wt, K, Kpad, HC);
        gemm_bf16_wmma<<<ggrid, 256, 0, stream>>>(hbuf, wt, xrout, NNODES, Kpad, HC);

        // segment softmax over incoming edges
        fill_i32<<<cdiv((long long)NNODES * H, T), T, 0, stream>>>(smax, (long long)NNODES * H, (int)0x80000000);
        fill_f32<<<cdiv((long long)NNODES * H, T), T, 0, stream>>>(den,  (long long)NNODES * H, 0.f);
        score_kernel<<<cdiv((long long)ETOT * H, 8), 256, 0, stream>>>(xl, xrout, ei, eaf, We, att, score, smax, H, Cc);
        softmax_edge_kernel<<<cdiv((long long)ETOT * H, T), T, 0, stream>>>(score, ei, smax, exb, den, H);

        // aggregate into xrout (xr no longer needed)
        fill_f32<<<cdiv((long long)NNODES * HC, T), T, 0, stream>>>(xrout, (long long)NNODES * HC, 0.f);
        aggregate_kernel<<<cdiv((long long)ETOT * HC / 4, T), T, 0, stream>>>(xl, ei, exb, den, xrout, H, Cc);

        // epilogue: +b (+ BN + ELU for layers 0..2) -> next h (stride HC, multiple of 64)
        if (i < 3) {
            const float* g  = (const float*)d_in[26 + 4 * i + 0];
            const float* be = (const float*)d_in[26 + 4 * i + 1];
            const float* mn = (const float*)d_in[26 + 4 * i + 2];
            const float* vr = (const float*)d_in[26 + 4 * i + 3];
            bias_bn_elu_kernel<<<cdiv((long long)NNODES * HC, T), T, 0, stream>>>(
                xrout, bb, g, be, mn, vr, hbuf, HC, 1);
        } else {
            bias_bn_elu_kernel<<<cdiv((long long)NNODES * HC, T), T, 0, stream>>>(
                xrout, bb, nullptr, nullptr, nullptr, nullptr, hbuf, HC, 0);
        }
    }

    final_lin_kernel<<<cdiv(NNODES, T), T, 0, stream>>>(hbuf, linW, linB, (float*)d_out);
}
